// Net_46213848105115
// MI455X (gfx1250) — matrix operand order repaired and checked by reading, not compiled, not run on hardware
//
#include <hip/hip_runtime.h>
#include <hip/hip_bf16.h>

// ---------------------------------------------------------------------------
// GAT (2-layer) for MI455X / gfx1250.
//  - GEMMs via v_wmma_f32_16x16x32_bf16, wave32, 16x64 strip per wave
//    (4 N-tiles register-blocked: one A fragment feeds 4 WMMAs).
//  - Edge softmax via monotone-uint atomicMax (segment max) + atomicAdd.
//  - Aggregation via float4 gathers + global_atomic_add_f32 scatters.
// ---------------------------------------------------------------------------

typedef __attribute__((ext_vector_type(8)))  float    v8f;
typedef __attribute__((ext_vector_type(16))) __bf16   bf16x16;
typedef __attribute__((ext_vector_type(4)))  unsigned u32x4;

union Frag { u32x4 u[2]; bf16x16 v; };

static __device__ __forceinline__ unsigned short f2bf(float f) {
  unsigned u = __float_as_uint(f);
  if ((u & 0x7fffffffu) > 0x7f800000u) return (unsigned short)((u >> 16) | 0x40u); // quiet NaN
  unsigned r = u + 0x7fffu + ((u >> 16) & 1u); // round-to-nearest-even
  return (unsigned short)(r >> 16);
}

// Monotone mapping float -> uint for atomicMax-based segment max.
static __device__ __forceinline__ unsigned fmap(float f) {
  unsigned b = __float_as_uint(f);
  return (b & 0x80000000u) ? ~b : (b | 0x80000000u);
}
static __device__ __forceinline__ float funmap(unsigned k) {
  return __uint_as_float((k & 0x80000000u) ? (k & 0x7fffffffu) : ~k);
}

// ---------------------------------------------------------------------------
// fp32 -> bf16 elementwise
__global__ void k_cvt_bf16(const float* __restrict__ in, unsigned short* __restrict__ out, int n) {
  int t = blockIdx.x * blockDim.x + threadIdx.x;
  if (t < n) out[t] = f2bf(in[t]);
}

// Fill u32 buffer (also used for f32 zero / mkey init)
__global__ void k_fill_u32(unsigned* __restrict__ p, unsigned v, int n) {
  int t = blockIdx.x * blockDim.x + threadIdx.x;
  if (t < n) p[t] = v;
}

// ---------------------------------------------------------------------------
// Pack W [K, Nld] (row-major f32) into per-lane B fragments (bf16):
// fragment (n_tile, ks): lane L holds col (L&15), K = ks*32 + ((L>>4)<<4) + j,
// j = 0..15 ascending (2 bf16 per dword).
__global__ void k_pack_B(const float* __restrict__ W, unsigned short* __restrict__ Bp,
                         int K, int Nld, int n_tiles) {
  int t = blockIdx.x * blockDim.x + threadIdx.x;     // one thread per (n_tile, ks, lane)
  int ksteps = K >> 5;
  int total = n_tiles * ksteps * 32;
  if (t >= total) return;
  int lane = t & 31;
  int rest = t >> 5;
  int ks = rest % ksteps;
  int n_tile = rest / ksteps;
  int col = lane & 15;
  int kb = ks * 32 + ((lane >> 4) << 4);             // 0 or 16 within the 32-K step
  int n = n_tile * 16 + col;
  unsigned short* o = Bp + (size_t)t * 16;
#pragma unroll
  for (int j = 0; j < 16; ++j) o[j] = f2bf(W[(size_t)(kb + j) * Nld + n]);
}

// ---------------------------------------------------------------------------
// bf16 WMMA GEMM: C[M, Nld] = A[M, K] (bf16 row-major) * Bpacked.
// One wave computes a 16x64 strip (4 consecutive N-tiles): a single A
// fragment per k-step is reused by 4 WMMAs, cutting load bytes/WMMA by 37.5%.
// Requires n_tiles % 4 == 0 (64 for GEMM1, 4 for GEMM2).
__global__ void k_gemm_wmma_bf16(const unsigned short* __restrict__ A,
                                 const unsigned short* __restrict__ Bp,
                                 float* __restrict__ C,
                                 int M, int Nld, int K, int n_tiles) {
  int wave = (blockIdx.x * blockDim.x + threadIdx.x) >> 5;
  int lane = threadIdx.x & 31;
  int m_tiles = M >> 4;
  int ngroups = n_tiles >> 2;
  if (wave >= m_tiles * ngroups) return;
  int m_tile = wave / ngroups;
  int ng = wave % ngroups;
  int ksteps = K >> 5;

  v8f c0 = {0.f, 0.f, 0.f, 0.f, 0.f, 0.f, 0.f, 0.f};
  v8f c1 = c0, c2 = c0, c3 = c0;

  // A fragment addressing (contiguous k order, same as packed B; not AMD's table order):
  //  lane<16:  row = lane,    K = {k0..k0+15}
  //  lane>=16: row = lane-16, K = {k0+16..k0+31}
  int row = m_tile * 16 + (lane & 15);
  int koff = (lane >> 4) << 4;                        // 0 or 16
  const unsigned short* arow = A + (size_t)row * K;
  // B fragments for n_tile = ng*4 + j ; fragment stride per n_tile = ksteps*512
  const size_t fstride = (size_t)ksteps * 32 * 16;    // ushorts per n_tile
  const unsigned short* b0p = Bp + (size_t)(ng * 4 + 0) * fstride + (size_t)lane * 16;
  const unsigned short* b1p = Bp + (size_t)(ng * 4 + 1) * fstride + (size_t)lane * 16;
  const unsigned short* b2p = Bp + (size_t)(ng * 4 + 2) * fstride + (size_t)lane * 16;
  const unsigned short* b3p = Bp + (size_t)(ng * 4 + 3) * fstride + (size_t)lane * 16;

  for (int ks = 0; ks < ksteps; ++ks) {
    int k0 = ks * 32 + koff;
    Frag a, b0, b1, b2, b3;
    a.u[0] = *(const u32x4*)(arow + k0);
    a.u[1] = *(const u32x4*)(arow + k0 + 8);
    size_t fo = (size_t)ks * 32 * 16;
    b0.u[0] = *(const u32x4*)(b0p + fo); b0.u[1] = *(const u32x4*)(b0p + fo + 8);
    b1.u[0] = *(const u32x4*)(b1p + fo); b1.u[1] = *(const u32x4*)(b1p + fo + 8);
    b2.u[0] = *(const u32x4*)(b2p + fo); b2.u[1] = *(const u32x4*)(b2p + fo + 8);
    b3.u[0] = *(const u32x4*)(b3p + fo); b3.u[1] = *(const u32x4*)(b3p + fo + 8);
    c0 = __builtin_amdgcn_wmma_f32_16x16x32_bf16(false, a.v, false, b0.v, (short)0, c0, false, false);
    c1 = __builtin_amdgcn_wmma_f32_16x16x32_bf16(false, a.v, false, b1.v, (short)0, c1, false, false);
    c2 = __builtin_amdgcn_wmma_f32_16x16x32_bf16(false, a.v, false, b2.v, (short)0, c2, false, false);
    c3 = __builtin_amdgcn_wmma_f32_16x16x32_bf16(false, a.v, false, b3.v, (short)0, c3, false, false);
  }

  // D layout: VGPR r: lanes 0-15 -> M=r, N=lane; lanes 16-31 -> M=r+8, N=lane-16.
  int ncol = (lane & 15);
  int mbase = m_tile * 16 + ((lane >> 4) << 3);
#pragma unroll
  for (int r = 0; r < 8; ++r) {
    float* crow = C + (size_t)(mbase + r) * Nld + (ng * 64 + ncol);
    crow[0]  = c0[r];
    crow[16] = c1[r];
    crow[32] = c2[r];
    crow[48] = c3[r];
  }
}

// ---------------------------------------------------------------------------
// Per-node attention dots: a_src[n,h] = <h[n,h,:], att_src[h,:]>, same for dst.
__global__ void k_node_dots(const float* __restrict__ Hf,
                            const float* __restrict__ att_s,
                            const float* __restrict__ att_d,
                            float* __restrict__ as_, float* __restrict__ ad_,
                            int N, int Hh, int Cc) {
  int t = blockIdx.x * blockDim.x + threadIdx.x;
  if (t >= N * Hh) return;
  int n = t / Hh, h = t % Hh;
  const float* hp = Hf + ((size_t)n * Hh + h) * Cc;
  const float* ws = att_s + (size_t)h * Cc;
  const float* wd = att_d + (size_t)h * Cc;
  float s = 0.f, d = 0.f;
  for (int c = 0; c < Cc; ++c) { float v = hp[c]; s += v * ws[c]; d += v * wd[c]; }
  as_[t] = s; ad_[t] = d;
}

// ---------------------------------------------------------------------------
// Edge logits (leaky relu) + segment max over dst via mapped atomicMax.
__global__ void k_edge_logits_max(const int* __restrict__ ei, int E, int N, int Hh,
                                  const float* __restrict__ as_, const float* __restrict__ ad_,
                                  float* __restrict__ elog, unsigned* __restrict__ mkey) {
  int t = blockIdx.x * blockDim.x + threadIdx.x;
  int Etot = E + N;
  if (t >= Etot * Hh) return;
  int e = t / Hh, h = t % Hh;
  int s = (e < E) ? ei[e] : (e - E);
  int d = (e < E) ? ei[E + e] : (e - E);
  float v = as_[s * Hh + h] + ad_[d * Hh + h];
  v = (v > 0.f) ? v : 0.2f * v;                 // leaky_relu, slope 0.2
  elog[t] = v;
  atomicMax(&mkey[d * Hh + h], fmap(v));
}

// exp(e - m[dst]) in-place + denom accumulation.
__global__ void k_edge_exp_den(const int* __restrict__ ei, int E, int N, int Hh,
                               const unsigned* __restrict__ mkey,
                               float* __restrict__ elog, float* __restrict__ den) {
  int t = blockIdx.x * blockDim.x + threadIdx.x;
  int Etot = E + N;
  if (t >= Etot * Hh) return;
  int e = t / Hh, h = t % Hh;
  int d = (e < E) ? ei[E + e] : (e - E);
  float m = funmap(mkey[d * Hh + h]);           // every node has a self loop -> finite
  float ex = expf(elog[t] - m);
  elog[t] = ex;
  atomicAdd(&den[d * Hh + h], ex);
}

// Weighted scatter-add aggregation: agg[dst,h,:] += h[src,h,:] * alpha.
// One thread per (edge, head, 4-float chunk).
__global__ void k_edge_aggregate(const int* __restrict__ ei, int E, int N, int Hh, int Cc,
                                 const float* __restrict__ Hf,
                                 const float* __restrict__ ex, const float* __restrict__ den,
                                 float* __restrict__ agg) {
  int vecs = Hh * (Cc >> 2);
  int t = blockIdx.x * blockDim.x + threadIdx.x;
  if (t >= (E + N) * vecs) return;
  int e = t / vecs;
  int rem = t % vecs;
  int h = rem / (Cc >> 2);
  int c4 = (rem % (Cc >> 2)) << 2;
  int s = (e < E) ? ei[e] : (e - E);
  int d = (e < E) ? ei[E + e] : (e - E);
  float alpha = ex[(size_t)e * Hh + h] / (den[(size_t)d * Hh + h] + 1e-16f);
  const float4 hv = *(const float4*)(Hf + ((size_t)s * Hh + h) * Cc + c4);
  float* o = agg + ((size_t)d * Hh + h) * Cc + c4;
  atomicAdd(o + 0, hv.x * alpha);
  atomicAdd(o + 1, hv.y * alpha);
  atomicAdd(o + 2, hv.z * alpha);
  atomicAdd(o + 3, hv.w * alpha);
}

// bias + ELU, emit bf16 (feeds layer-2 WMMA GEMM).
__global__ void k_bias_elu_bf16(const float* __restrict__ agg, const float* __restrict__ b,
                                unsigned short* __restrict__ out, int N, int F) {
  int t = blockIdx.x * blockDim.x + threadIdx.x;
  if (t >= N * F) return;
  float v = agg[t] + b[t % F];
  v = (v > 0.f) ? v : (expf(v) - 1.f);          // elu
  out[t] = f2bf(v);
}

// Final: out = log_softmax(agg + b, axis=-1); 64 cols, one wave32 per node.
__global__ void k_bias_logsoftmax64(const float* __restrict__ agg, const float* __restrict__ b,
                                    float* __restrict__ out, int N) {
  int wave = (blockIdx.x * blockDim.x + threadIdx.x) >> 5;
  int lane = threadIdx.x & 31;
  if (wave >= N) return;
  const float* row = agg + (size_t)wave * 64;
  float v0 = row[lane] + b[lane];
  float v1 = row[lane + 32] + b[lane + 32];
  float mx = fmaxf(v0, v1);
#pragma unroll
  for (int o = 16; o > 0; o >>= 1) mx = fmaxf(mx, __shfl_xor(mx, o, 32));
  float s = expf(v0 - mx) + expf(v1 - mx);
#pragma unroll
  for (int o = 16; o > 0; o >>= 1) s += __shfl_xor(s, o, 32);
  float ls = logf(s);
  out[(size_t)wave * 64 + lane] = v0 - mx - ls;
  out[(size_t)wave * 64 + lane + 32] = v1 - mx - ls;
}

// ---------------------------------------------------------------------------
static inline int cdiv(long long a, long long b) { return (int)((a + b - 1) / b); }

extern "C" void kernel_launch(void* const* d_in, const int* in_sizes, int n_in,
                              void* d_out, int out_size, void* d_ws, size_t ws_size,
                              hipStream_t stream) {
  const float* x        = (const float*)d_in[0];
  const int*   ei       = (const int*)  d_in[1];
  const float* W1       = (const float*)d_in[2];
  const float* att_src1 = (const float*)d_in[3];
  const float* att_dst1 = (const float*)d_in[4];
  const float* b1       = (const float*)d_in[5];
  const float* W2       = (const float*)d_in[6];
  const float* att_src2 = (const float*)d_in[7];
  const float* att_dst2 = (const float*)d_in[8];
  const float* b2       = (const float*)d_in[9];

  const int F_IN = 256, H1 = 8, C1 = 128, F1 = H1 * C1 /*1024*/, OUT = 64;
  const int N = in_sizes[0] / F_IN;          // 30000
  const int E = in_sizes[1] / 2;             // 300000
  const int Etot = E + N;                    // self loops appended
  const int NT1 = F1 / 16;                   // 64 n-tiles, GEMM1
  const int NT2 = OUT / 16;                  // 4  n-tiles, GEMM2

  // ---- workspace layout (256B aligned) ----
  char* base = (char*)d_ws;
  size_t off = 0;
  auto alloc = [&](size_t bytes) -> char* {
    char* p = base + off;
    off += (bytes + 255) & ~(size_t)255;
    return p;
  };
  unsigned short* x16   = (unsigned short*)alloc((size_t)N * F_IN * 2);
  unsigned short* w1p   = (unsigned short*)alloc((size_t)NT1 * (F_IN / 32) * 32 * 16 * 2);
  unsigned short* w2p   = (unsigned short*)alloc((size_t)NT2 * (F1 / 32) * 32 * 16 * 2);
  float*    asrc1 = (float*)   alloc((size_t)N * H1 * 4);
  float*    adst1 = (float*)   alloc((size_t)N * H1 * 4);
  float*    e1    = (float*)   alloc((size_t)Etot * H1 * 4);
  unsigned* mkey1 = (unsigned*)alloc((size_t)N * H1 * 4);
  float*    den1  = (float*)   alloc((size_t)N * H1 * 4);
  float*    asrc2 = (float*)   alloc((size_t)N * 4);
  float*    adst2 = (float*)   alloc((size_t)N * 4);
  float*    e2    = (float*)   alloc((size_t)Etot * 4);
  unsigned* mkey2 = (unsigned*)alloc((size_t)N * 4);
  float*    den2  = (float*)   alloc((size_t)N * 4);
  float*    h2    = (float*)   alloc((size_t)N * OUT * 4);
  float*    agg2  = (float*)   alloc((size_t)N * OUT * 4);
  float*    h1    = (float*)   alloc((size_t)N * F1 * 4);   // reused as h1b (bf16) after layer-1 agg
  float*    agg1  = (float*)   alloc((size_t)N * F1 * 4);
  unsigned short* h1b = (unsigned short*)h1;                // alias: h1 dead by ELU time

  const int B = 256;

  // ---- prep: conversions & weight packing ----
  k_cvt_bf16<<<cdiv((long long)N * F_IN, B), B, 0, stream>>>(x, x16, N * F_IN);
  k_pack_B<<<cdiv(NT1 * (F_IN / 32) * 32, B), B, 0, stream>>>(W1, w1p, F_IN, F1, NT1);
  k_pack_B<<<cdiv(NT2 * (F1 / 32) * 32, B), B, 0, stream>>>(W2, w2p, F1, OUT, NT2);

  // ---- layer 1: GEMM (WMMA bf16, 16x64 strip per wave) ----
  {
    int strips = (N / 16) * (NT1 / 4);
    k_gemm_wmma_bf16<<<cdiv(strips, 8), B, 0, stream>>>(x16, w1p, h1, N, F1, F_IN, NT1);
  }
  k_node_dots<<<cdiv((long long)N * H1, B), B, 0, stream>>>(h1, att_src1, att_dst1,
                                                            asrc1, adst1, N, H1, C1);
  // init accumulators (fresh every call)
  k_fill_u32<<<cdiv((long long)N * H1, B), B, 0, stream>>>(mkey1, 0u, N * H1);
  k_fill_u32<<<cdiv((long long)N * H1, B), B, 0, stream>>>((unsigned*)den1, 0u, N * H1);
  k_fill_u32<<<cdiv((long long)N * F1, B), B, 0, stream>>>((unsigned*)agg1, 0u, N * F1);

  k_edge_logits_max<<<cdiv((long long)Etot * H1, B), B, 0, stream>>>(ei, E, N, H1,
                                                                     asrc1, adst1, e1, mkey1);
  k_edge_exp_den<<<cdiv((long long)Etot * H1, B), B, 0, stream>>>(ei, E, N, H1, mkey1, e1, den1);
  k_edge_aggregate<<<cdiv((long long)Etot * H1 * (C1 / 4), B), B, 0, stream>>>(
      ei, E, N, H1, C1, h1, e1, den1, agg1);

  k_bias_elu_bf16<<<cdiv((long long)N * F1, B), B, 0, stream>>>(agg1, b1, h1b, N, F1);

  // ---- layer 2: GEMM (WMMA bf16, whole 64-wide N extent per wave) ----
  {
    int strips = (N / 16) * (NT2 / 4);
    k_gemm_wmma_bf16<<<cdiv(strips, 8), B, 0, stream>>>(h1b, w2p, h2, N, OUT, F1, NT2);
  }
  k_node_dots<<<cdiv((long long)N, B), B, 0, stream>>>(h2, att_src2, att_dst2,
                                                       asrc2, adst2, N, 1, OUT);
  k_fill_u32<<<cdiv((long long)N, B), B, 0, stream>>>(mkey2, 0u, N);
  k_fill_u32<<<cdiv((long long)N, B), B, 0, stream>>>((unsigned*)den2, 0u, N);
  k_fill_u32<<<cdiv((long long)N * OUT, B), B, 0, stream>>>((unsigned*)agg2, 0u, N * OUT);

  k_edge_logits_max<<<cdiv((long long)Etot, B), B, 0, stream>>>(ei, E, N, 1,
                                                                asrc2, adst2, e2, mkey2);
  k_edge_exp_den<<<cdiv((long long)Etot, B), B, 0, stream>>>(ei, E, N, 1, mkey2, e2, den2);
  k_edge_aggregate<<<cdiv((long long)Etot * (OUT / 4), B), B, 0, stream>>>(
      ei, E, N, 1, OUT, h2, e2, den2, agg2);

  // heads==1, concat=False -> mean over 1 head is identity; add bias + log_softmax
  k_bias_logsoftmax64<<<cdiv((long long)N * 32, B), B, 0, stream>>>(agg2, b2, (float*)d_out, N);
}
